// GATNet_79809082294962
// MI455X (gfx1250) — matrix operand order repaired and checked by reading, not compiled, not run on hardware
//
#include <hip/hip_runtime.h>
#include <math.h>

#define NN 20000
#define EE 640000
#define BB 64
#define ET_TOT (EE + NN)
#define NEG_SLOPE 0.2f
#define EPS_BN 1e-5f

typedef __attribute__((ext_vector_type(16))) _Float16 v16h;
typedef __attribute__((ext_vector_type(8)))  float    v8f;
typedef __attribute__((ext_vector_type(8)))  unsigned v8u;

static inline int cdiv(int a, int b) { return (a + b - 1) / b; }

// ---------- device helpers ----------
__device__ __forceinline__ unsigned f2ord(float f) {
  unsigned u = __float_as_uint(f);
  return (u & 0x80000000u) ? ~u : (u | 0x80000000u);
}
__device__ __forceinline__ float ord2f(unsigned u) {
  return __uint_as_float((u & 0x80000000u) ? (u & 0x7fffffffu) : ~u);
}
__device__ __forceinline__ float sigmoidf_(float x) { return 1.f / (1.f + __expf(-x)); }

__device__ __forceinline__ void edge_sd(const int* ei, int e, int& s, int& d) {
  if (e < EE) { s = ei[e]; d = ei[EE + e]; } else { s = e - EE; d = s; }
}
__device__ __forceinline__ unsigned pack2h(float f0, float f1) {
  _Float16 h0 = (_Float16)f0, h1 = (_Float16)f1;
  return ((unsigned)__builtin_bit_cast(unsigned short, h1) << 16) |
         (unsigned)__builtin_bit_cast(unsigned short, h0);
}

// ---------- self-loop edge-attr (mean of incoming attrs per node) ----------
__global__ void k_loop_acc(const int* __restrict__ ei, const float* __restrict__ eattr,
                           float* __restrict__ cnt, float* __restrict__ lsum) {
  for (int e = blockIdx.x * blockDim.x + threadIdx.x; e < EE; e += gridDim.x * blockDim.x) {
    int d = ei[EE + e];
    atomicAdd(&cnt[d], 1.f);
#pragma unroll
    for (int k = 0; k < 8; ++k) atomicAdd(&lsum[(size_t)d * 8 + k], eattr[(size_t)e * 8 + k]);
  }
}
__global__ void k_loop_fin(float* __restrict__ lsum, const float* __restrict__ cnt) {
  for (int i = blockIdx.x * blockDim.x + threadIdx.x; i < NN * 8; i += gridDim.x * blockDim.x)
    lsum[i] /= fmaxf(cnt[i >> 3], 1.f);
}

// ---------- WMMA GEMM: Y[M,Cout] = X[M,K] @ W[K,Cout] + bias ----------
// Block = 256 thr = 8 waves covering 64 rows x 32 cols.  K (compile-time) tiled by 32.
// LDS holds f16 pairs packed in u32 (pitch 20 u32 = 80B, rows 16B aligned).
// B stored transposed so each lane's 16 B-values are contiguous (2x ds_load_b128).
// Fragment pair indices, contiguous k = 16*khigh + s for both A and B (wave32):
//   A lane(l): row m=l&15, khigh=l>>4, pairs {8*khigh+q}, q=0..7
//   B lane(l): col n=l&15, khigh=l>>4, pairs {8*khigh+q}, q=0..7
template <int K>
__global__ __launch_bounds__(256)
void k_gemm_bias(const float* __restrict__ X, const float* __restrict__ W,
                 const float* __restrict__ bias, float* __restrict__ Y,
                 int M, int Cout) {
  __shared__ __align__(16) unsigned lA[64][20];
  __shared__ __align__(16) unsigned lBt[32][20];
  const int tid = threadIdx.x;
  const int lane = tid & 31, wave = tid >> 5;
  const int wm = wave >> 1, wn = wave & 1;
  const int row0 = blockIdx.x * 64, col0 = blockIdx.y * 32;
  const bool interior = (row0 + 64 <= M);
  v8f acc = {};
#pragma unroll
  for (int k0 = 0; k0 < K; k0 += 32) {
    // A tile: 64 rows x 16 u32-pairs (1024), 256 threads -> 4 each
    if (interior) {
#pragma unroll
      for (int i = 0; i < 4; ++i) {
        int idx = tid + i * 256;
        int r = idx >> 4, p = idx & 15;
        const float* px = X + (size_t)(row0 + r) * K + k0 + 2 * p;
        lA[r][p] = pack2h(px[0], px[1]);
      }
    } else {
#pragma unroll
      for (int i = 0; i < 4; ++i) {
        int idx = tid + i * 256;
        int r = idx >> 4, p = idx & 15;
        int gr = row0 + r;
        float f0 = 0.f, f1 = 0.f;
        if (gr < M) {
          const float* px = X + (size_t)gr * K + k0 + 2 * p;
          f0 = px[0]; f1 = px[1];
        }
        lA[r][p] = pack2h(f0, f1);
      }
    }
    // B tile transposed: 32 cols x 16 u32-pairs (512), 256 threads -> 2 each
#pragma unroll
    for (int i = 0; i < 2; ++i) {
      int idx = tid + i * 256;
      int c = idx >> 4, p = idx & 15;
      const float* pw = W + (size_t)(k0 + 2 * p) * Cout + col0 + c;
      lBt[c][p] = pack2h(pw[0], pw[Cout]);
    }
    __syncthreads();
    const int mrow = 16 * wm + (lane & 15);
    const int half = lane >> 4;
    v8u au, bu;
#pragma unroll
    for (int q = 0; q < 8; ++q) au[q] = lA[mrow][8 * half + q];
    const int ncol = 16 * wn + (lane & 15);
#pragma unroll
    for (int q = 0; q < 8; ++q) bu[q] = lBt[ncol][8 * half + q];
    v16h a = __builtin_bit_cast(v16h, au);
    v16h b = __builtin_bit_cast(v16h, bu);
    acc = __builtin_amdgcn_wmma_f32_16x16x32_f16(false, a, false, b, (short)0, acc, false, false);
    __syncthreads();
  }
  const int gcol = col0 + 16 * wn + (lane & 15);
  const int grow0 = row0 + 16 * wm + 8 * (lane >> 4);
  const float bv = bias[gcol];
  float* yp = Y + (size_t)grow0 * Cout + gcol;
  if (grow0 + 8 <= M) {
#pragma unroll
    for (int vv = 0; vv < 8; ++vv) yp[(size_t)vv * Cout] = acc[vv] + bv;
  } else {
#pragma unroll
    for (int vv = 0; vv < 8; ++vv)
      if (grow0 + vv < M) yp[(size_t)vv * Cout] = acc[vv] + bv;
  }
}

// ---------- edge pass 1: logits + per-(dst,head) max (wave per edge) ----------
// Lane owns features f = VEC*lane + t (VEC = F/32): one float4/float2 per row read.
// Head group = G = C/VEC lanes; butterfly-reduce within group; lane (h*G) writes head h.
template <int F, int C, int H>
__global__ __launch_bounds__(256)
void k_edge_logit(const int* __restrict__ ei, const float* __restrict__ eattr,
                  const float* __restrict__ lattr,
                  const float* __restrict__ xl, const float* __restrict__ xr,
                  const float* __restrict__ We, const float* __restrict__ att,
                  float* __restrict__ elog, unsigned* __restrict__ nmax) {
  constexpr int VEC = F / 32;
  constexpr int G = C / VEC;
  __shared__ float sWe[8 * F];
  __shared__ float sAtt[F];
  for (int i = threadIdx.x; i < 8 * F; i += blockDim.x) sWe[i] = We[i];
  for (int i = threadIdx.x; i < F; i += blockDim.x) sAtt[i] = att[i];
  __syncthreads();
  const int lane = threadIdx.x & 31;
  const int e = blockIdx.x * (blockDim.x >> 5) + (threadIdx.x >> 5);
  if (e >= ET_TOT) return;
  int s, d; edge_sd(ei, e, s, d);
  const float* ea = (e < EE) ? (eattr + (size_t)e * 8) : (lattr + (size_t)(e - EE) * 8);
  float av[8];
#pragma unroll
  for (int k = 0; k < 8; ++k) av[k] = ea[k];
  const float* pxl = xl + (size_t)s * F + VEC * lane;
  const float* pxr = xr + (size_t)d * F + VEC * lane;
  __builtin_prefetch(pxl, 0, 1);
  __builtin_prefetch(pxr, 0, 1);
  float xv[VEC], rv[VEC];
  if constexpr (VEC == 4) {
    float4 a4 = *(const float4*)pxl; xv[0] = a4.x; xv[1] = a4.y; xv[2] = a4.z; xv[3] = a4.w;
    float4 b4 = *(const float4*)pxr; rv[0] = b4.x; rv[1] = b4.y; rv[2] = b4.z; rv[3] = b4.w;
  } else {
    float2 a2 = *(const float2*)pxl; xv[0] = a2.x; xv[1] = a2.y;
    float2 b2 = *(const float2*)pxr; rv[0] = b2.x; rv[1] = b2.y;
  }
  float acc = 0.f;
#pragma unroll
  for (int t = 0; t < VEC; ++t) {
    int f = VEC * lane + t;
    float em = 0.f;
#pragma unroll
    for (int k = 0; k < 8; ++k) em += av[k] * sWe[k * F + f];
    float m = xv[t] + rv[t] + em;
    m = (m > 0.f) ? m : NEG_SLOPE * m;
    acc += m * sAtt[f];
  }
#pragma unroll
  for (int off = G / 2; off > 0; off >>= 1) acc += __shfl_xor(acc, off, 32);
  if ((lane & (G - 1)) == 0) {
    int h = lane / G;
    elog[(size_t)e * H + h] = acc;
    atomicMax(&nmax[(size_t)d * H + h], f2ord(acc));
  }
}

// ---------- edge pass 2: ex = exp(logit - max), denominator ----------
__global__ void k_edge_den(const int* __restrict__ ei, float* __restrict__ elog,
                           const unsigned* __restrict__ nmax, float* __restrict__ nden, int H) {
  const int n = ET_TOT * H;
  for (int i = blockIdx.x * blockDim.x + threadIdx.x; i < n; i += gridDim.x * blockDim.x) {
    int e = i / H, h = i - e * H;
    int s, d; edge_sd(ei, e, s, d); (void)s;
    unsigned u = nmax[(size_t)d * H + h];
    float mx = u ? ord2f(u) : 0.f;
    float ex = __expf(elog[i] - mx);
    elog[i] = ex;
    atomicAdd(&nden[(size_t)d * H + h], ex);
  }
}

// ---------- edge pass 3: agg[dst] += alpha * xl[src] (wave per edge) ----------
template <int F, int C, int H>
__global__ __launch_bounds__(256)
void k_edge_scatter(const int* __restrict__ ei, const float* __restrict__ xl,
                    const float* __restrict__ elog, const float* __restrict__ nden,
                    float* __restrict__ agg) {
  constexpr int VEC = F / 32;
  constexpr int G = C / VEC;
  const int lane = threadIdx.x & 31;
  const int e = blockIdx.x * (blockDim.x >> 5) + (threadIdx.x >> 5);
  if (e >= ET_TOT) return;
  int s, d; edge_sd(ei, e, s, d);
  const int h = lane / G;
  const float al = elog[(size_t)e * H + h] / fmaxf(nden[(size_t)d * H + h], 1e-16f);
  const float* pxl = xl + (size_t)s * F + VEC * lane;
  float* pagg = agg + (size_t)d * F + VEC * lane;
  __builtin_prefetch(pxl, 0, 1);
  if constexpr (VEC == 4) {
    float4 v = *(const float4*)pxl;
    atomicAdd(&pagg[0], al * v.x);
    atomicAdd(&pagg[1], al * v.y);
    atomicAdd(&pagg[2], al * v.z);
    atomicAdd(&pagg[3], al * v.w);
  } else {
    float2 v = *(const float2*)pxl;
    atomicAdd(&pagg[0], al * v.x);
    atomicAdd(&pagg[1], al * v.y);
  }
}

// ---------- node finalize: h = elu(agg + bias) ----------
__global__ void k_node_fin(float* __restrict__ h, const float* __restrict__ bias, int F) {
  const int n = NN * F;
  for (int i = blockIdx.x * blockDim.x + threadIdx.x; i < n; i += gridDim.x * blockDim.x) {
    float v = h[i] + bias[i % F];
    h[i] = (v > 0.f) ? v : expm1f(v);
  }
}

// ---------- batchnorm ----------
__global__ __launch_bounds__(256)
void k_bn_stats(const float* __restrict__ h, float* __restrict__ mu, float* __restrict__ var, int F) {
  __shared__ float ssum[256], ssq[256];
  const int c = blockIdx.x;
  float s = 0.f, q = 0.f;
  for (int n = threadIdx.x; n < NN; n += blockDim.x) {
    float x = h[(size_t)n * F + c];
    s += x; q += x * x;
  }
  ssum[threadIdx.x] = s; ssq[threadIdx.x] = q;
  __syncthreads();
  for (int o = 128; o > 0; o >>= 1) {
    if ((int)threadIdx.x < o) { ssum[threadIdx.x] += ssum[threadIdx.x + o]; ssq[threadIdx.x] += ssq[threadIdx.x + o]; }
    __syncthreads();
  }
  if (threadIdx.x == 0) {
    float m = ssum[0] / (float)NN;
    mu[c] = m;
    var[c] = ssq[0] / (float)NN - m * m;
  }
}
__global__ void k_bn_apply(float* __restrict__ h, const float* __restrict__ mu,
                           const float* __restrict__ var, const float* __restrict__ gamma,
                           const float* __restrict__ beta, int F) {
  const int n = NN * F;
  for (int i = blockIdx.x * blockDim.x + threadIdx.x; i < n; i += gridDim.x * blockDim.x) {
    int c = i % F;
    h[i] = (h[i] - mu[c]) * rsqrtf(var[c] + EPS_BN) * gamma[c] + beta[c];
  }
}

// ---------- pooling ----------
__global__ void k_gcnt(const int* __restrict__ batch, float* __restrict__ gcnt) {
  for (int n = blockIdx.x * blockDim.x + threadIdx.x; n < NN; n += gridDim.x * blockDim.x)
    atomicAdd(&gcnt[batch[n]], 1.f);
}
__global__ void k_pool_sums(const float* __restrict__ h, const int* __restrict__ batch,
                            float* __restrict__ xsum, unsigned* __restrict__ xmaxb) {
  const int n = NN * 64;
  for (int i = blockIdx.x * blockDim.x + threadIdx.x; i < n; i += gridDim.x * blockDim.x) {
    int nd = i >> 6, c = i & 63;
    int b = batch[nd];
    float v = h[i];
    atomicAdd(&xsum[b * 64 + c], v);
    atomicMax(&xmaxb[b * 64 + c], f2ord(v));
  }
}
__global__ void k_pool_fin(const float* __restrict__ xsum, const float* __restrict__ gcnt,
                           float* __restrict__ xmean, unsigned* __restrict__ xmaxb) {
  for (int i = blockIdx.x * blockDim.x + threadIdx.x; i < BB * 64; i += gridDim.x * blockDim.x) {
    int b = i >> 6;
    xmean[i] = xsum[i] / fmaxf(gcnt[b], 1.f);
    unsigned u = xmaxb[i];
    ((float*)xmaxb)[i] = u ? ord2f(u) : 0.f;
  }
}
__global__ void k_gate_logit(const float* __restrict__ h, const float* __restrict__ gW,
                             const float* __restrict__ gb, const int* __restrict__ batch,
                             float* __restrict__ glog, unsigned* __restrict__ gmaxb) {
  for (int n = blockIdx.x * blockDim.x + threadIdx.x; n < NN; n += gridDim.x * blockDim.x) {
    float s = gb[0];
#pragma unroll
    for (int c = 0; c < 64; ++c) s += h[(size_t)n * 64 + c] * gW[c];
    glog[n] = s;
    atomicMax(&gmaxb[batch[n]], f2ord(s));
  }
}
__global__ void k_seg_exp(float* __restrict__ lg, const unsigned* __restrict__ mx,
                          float* __restrict__ den, const int* __restrict__ batch) {
  for (int n = blockIdx.x * blockDim.x + threadIdx.x; n < NN; n += gridDim.x * blockDim.x) {
    int b = batch[n];
    unsigned u = mx[b];
    float m = u ? ord2f(u) : 0.f;
    float ex = __expf(lg[n] - m);
    lg[n] = ex;
    atomicAdd(&den[b], ex);
  }
}
__global__ void k_seg_scatter64(const float* __restrict__ h, const float* __restrict__ ex,
                                const float* __restrict__ den, const int* __restrict__ batch,
                                float* __restrict__ out) {
  for (int n = blockIdx.x * blockDim.x + threadIdx.x; n < NN; n += gridDim.x * blockDim.x) {
    int b = batch[n];
    float a = ex[n] / fmaxf(den[b], 1e-16f);
#pragma unroll
    for (int c = 0; c < 64; ++c) atomicAdd(&out[b * 64 + c], a * h[(size_t)n * 64 + c]);
  }
}

// ---------- Set2Set ----------
__global__ void k_lstm_gates(const float* __restrict__ qstar, const float* __restrict__ hs,
                             const float* __restrict__ Wih, const float* __restrict__ Whh,
                             const float* __restrict__ bih, const float* __restrict__ bhh,
                             float* __restrict__ gates) {
  const int n = 256 * BB;
  for (int i = blockIdx.x * blockDim.x + threadIdx.x; i < n; i += gridDim.x * blockDim.x) {
    int b = i % BB, r = i / BB;
    float g = bih[r] + bhh[r];
    for (int k = 0; k < 128; ++k) g += qstar[(size_t)b * 128 + k] * Wih[(size_t)r * 128 + k];
    for (int k = 0; k < 64; ++k) g += hs[(size_t)b * 64 + k] * Whh[(size_t)r * 64 + k];
    gates[i] = g;  // layout [r][b]
  }
}
__global__ void k_lstm_update(const float* __restrict__ gates, float* __restrict__ cs,
                              float* __restrict__ hs) {
  for (int i = blockIdx.x * blockDim.x + threadIdx.x; i < BB * 64; i += gridDim.x * blockDim.x) {
    int b = i % BB, j = i / BB;
    float gi = gates[(j) * BB + b];
    float gf = gates[(64 + j) * BB + b];
    float gg = gates[(128 + j) * BB + b];
    float go = gates[(192 + j) * BB + b];
    float c = sigmoidf_(gf) * cs[b * 64 + j] + sigmoidf_(gi) * tanhf(gg);
    cs[b * 64 + j] = c;
    hs[b * 64 + j] = sigmoidf_(go) * tanhf(c);
  }
}
__global__ void k_s2s_logit(const float* __restrict__ h, const float* __restrict__ hs,
                            const int* __restrict__ batch, float* __restrict__ slog,
                            unsigned* __restrict__ smaxb) {
  for (int n = blockIdx.x * blockDim.x + threadIdx.x; n < NN; n += gridDim.x * blockDim.x) {
    int b = batch[n];
    float s = 0.f;
#pragma unroll
    for (int c = 0; c < 64; ++c) s += h[(size_t)n * 64 + c] * hs[b * 64 + c];
    slog[n] = s;
    atomicMax(&smaxb[b], f2ord(s));
  }
}
__global__ void k_qstar(const float* __restrict__ hs, const float* __restrict__ rr,
                        float* __restrict__ qstar) {
  for (int i = blockIdx.x * blockDim.x + threadIdx.x; i < BB * 128; i += gridDim.x * blockDim.x) {
    int b = i >> 7, c = i & 127;
    qstar[i] = (c < 64) ? hs[b * 64 + c] : rr[b * 64 + c - 64];
  }
}
__global__ void k_concat(const float* __restrict__ xsum, const float* __restrict__ xmaxf,
                         const float* __restrict__ xmean, const float* __restrict__ xatt,
                         const float* __restrict__ qstar, float* __restrict__ z) {
  for (int i = blockIdx.x * blockDim.x + threadIdx.x; i < BB * 384; i += gridDim.x * blockDim.x) {
    int b = i / 384, c = i - b * 384;
    float v;
    if (c < 64) v = xsum[b * 64 + c];
    else if (c < 128) v = xmaxf[b * 64 + c - 64];
    else if (c < 192) v = xmean[b * 64 + c - 128];
    else if (c < 256) v = xatt[b * 64 + c - 192];
    else v = qstar[b * 128 + c - 256];
    z[i] = v;
  }
}
__global__ void k_mlp(const float* __restrict__ in, const float* __restrict__ W,
                      const float* __restrict__ bias, float* __restrict__ out,
                      int In, int Out, int doRelu) {
  for (int i = blockIdx.x * blockDim.x + threadIdx.x; i < BB * Out; i += gridDim.x * blockDim.x) {
    int b = i / Out, o = i - b * Out;
    float s = bias[o];
    for (int k = 0; k < In; ++k) s += in[(size_t)b * In + k] * W[(size_t)k * Out + o];
    out[i] = doRelu ? fmaxf(s, 0.f) : s;
  }
}

// ---------- host-side layer runner ----------
static void run_gat_layer(const float* hin, float* hout, int F, int H,
                          const float* Wl, const float* bl, const float* Wr, const float* br,
                          const float* We, const float* att, const float* bias,
                          const float* gamma, const float* beta,
                          const int* ei, const float* eattr, const float* lattr,
                          float* xl, float* xr, float* elog, unsigned* nmax, float* nden,
                          float* mu, float* var, hipStream_t stream) {
  dim3 g(cdiv(NN, 64), F / 32);
  k_gemm_bias<128><<<g, 256, 0, stream>>>(hin, Wl, bl, xl, NN, F);
  k_gemm_bias<128><<<g, 256, 0, stream>>>(hin, Wr, br, xr, NN, F);
  hipMemsetAsync(nmax, 0, (size_t)NN * H * 4, stream);
  hipMemsetAsync(nden, 0, (size_t)NN * H * 4, stream);
  hipMemsetAsync(hout, 0, (size_t)NN * F * 4, stream);
  const int wgrid = cdiv(ET_TOT, 8);
  if (H == 4) {
    k_edge_logit<128, 32, 4><<<wgrid, 256, 0, stream>>>(ei, eattr, lattr, xl, xr, We, att, elog, nmax);
    k_edge_den<<<cdiv(ET_TOT * 4, 256), 256, 0, stream>>>(ei, elog, nmax, nden, 4);
    k_edge_scatter<128, 32, 4><<<wgrid, 256, 0, stream>>>(ei, xl, elog, nden, hout);
  } else {
    k_edge_logit<64, 64, 1><<<wgrid, 256, 0, stream>>>(ei, eattr, lattr, xl, xr, We, att, elog, nmax);
    k_edge_den<<<cdiv(ET_TOT, 256), 256, 0, stream>>>(ei, elog, nmax, nden, 1);
    k_edge_scatter<64, 64, 1><<<wgrid, 256, 0, stream>>>(ei, xl, elog, nden, hout);
  }
  k_node_fin<<<cdiv(NN * F, 256), 256, 0, stream>>>(hout, bias, F);
  k_bn_stats<<<F, 256, 0, stream>>>(hout, mu, var, F);
  k_bn_apply<<<cdiv(NN * F, 256), 256, 0, stream>>>(hout, mu, var, gamma, beta, F);
}

extern "C" void kernel_launch(void* const* d_in, const int* in_sizes, int n_in,
                              void* d_out, int out_size, void* d_ws, size_t ws_size,
                              hipStream_t stream) {
  (void)in_sizes; (void)n_in; (void)out_size; (void)ws_size;
  const float* x     = (const float*)d_in[0];
  const int*   ei    = (const int*)d_in[1];
  const float* eattr = (const float*)d_in[2];
  const int*   batch = (const int*)d_in[3];
  auto F32 = [&](int i) { return (const float*)d_in[i]; };
  const float* gateW = F32(31);
  const float* gateB = F32(32);
  const float* Wih = F32(33);
  const float* Whh = F32(34);
  const float* bih = F32(35);
  const float* bhh = F32(36);
  const float* mW[4] = {F32(37), F32(39), F32(41), F32(43)};
  const float* mB[4] = {F32(38), F32(40), F32(42), F32(44)};

  // workspace carve (256B aligned)
  char* w = (char*)d_ws;
  auto alloc = [&](size_t nf) { float* p = (float*)w; w += ((nf * 4 + 255) / 256) * 256; return p; };
  float* xl    = alloc((size_t)NN * 128);
  float* xr    = alloc((size_t)NN * 128);
  float* hA    = alloc((size_t)NN * 128);
  float* hB    = alloc((size_t)NN * 128);
  float* lattr = alloc((size_t)NN * 8);
  float* cnt   = alloc(NN);
  float* elog  = alloc((size_t)ET_TOT * 4);
  unsigned* nmax = (unsigned*)alloc((size_t)NN * 4);
  float* nden  = alloc((size_t)NN * 4);
  float* mu    = alloc(128);
  float* var   = alloc(128);
  float* gcnt  = alloc(BB);
  float* xsum  = alloc(BB * 64);
  unsigned* xmaxb = (unsigned*)alloc(BB * 64);
  float* xmean = alloc(BB * 64);
  float* xatt  = alloc(BB * 64);
  float* glog  = alloc(NN);
  unsigned* gmaxb = (unsigned*)alloc(BB);
  float* gden  = alloc(BB);
  float* qstar = alloc(BB * 128);
  float* hs    = alloc(BB * 64);
  float* cs    = alloc(BB * 64);
  float* rr    = alloc(BB * 64);
  float* slog  = alloc(NN);
  unsigned* smaxb = (unsigned*)alloc(BB);
  float* sden  = alloc(BB);
  float* gates = alloc(256 * BB);
  float* z     = alloc(BB * 384);
  float* t1    = alloc(BB * 128);
  float* t2    = alloc(BB * 64);
  float* t3    = alloc(BB * 32);

  // self-loop attrs
  hipMemsetAsync(cnt, 0, (size_t)NN * 4, stream);
  hipMemsetAsync(lattr, 0, (size_t)NN * 8 * 4, stream);
  k_loop_acc<<<cdiv(EE, 256), 256, 0, stream>>>(ei, eattr, cnt, lattr);
  k_loop_fin<<<cdiv(NN * 8, 256), 256, 0, stream>>>(lattr, cnt);

  // GATv2 layers
  auto L = [&](int l, int p) { return F32(4 + l * 9 + p); };
  run_gat_layer(x,  hA, 128, 4, L(0,0), L(0,1), L(0,2), L(0,3), L(0,4), L(0,5), L(0,6), L(0,7), L(0,8),
                ei, eattr, lattr, xl, xr, elog, nmax, nden, mu, var, stream);
  run_gat_layer(hA, hB, 128, 4, L(1,0), L(1,1), L(1,2), L(1,3), L(1,4), L(1,5), L(1,6), L(1,7), L(1,8),
                ei, eattr, lattr, xl, xr, elog, nmax, nden, mu, var, stream);
  run_gat_layer(hB, hA,  64, 1, L(2,0), L(2,1), L(2,2), L(2,3), L(2,4), L(2,5), L(2,6), L(2,7), L(2,8),
                ei, eattr, lattr, xl, xr, elog, nmax, nden, mu, var, stream);
  float* h = hA;  // final node features [NN,64]

  // pooling: sum / max / mean / attentional
  hipMemsetAsync(gcnt, 0, BB * 4, stream);
  hipMemsetAsync(xsum, 0, BB * 64 * 4, stream);
  hipMemsetAsync(xmaxb, 0, BB * 64 * 4, stream);
  hipMemsetAsync(xatt, 0, BB * 64 * 4, stream);
  hipMemsetAsync(gmaxb, 0, BB * 4, stream);
  hipMemsetAsync(gden, 0, BB * 4, stream);
  k_gcnt<<<cdiv(NN, 256), 256, 0, stream>>>(batch, gcnt);
  k_pool_sums<<<cdiv(NN * 64, 256), 256, 0, stream>>>(h, batch, xsum, xmaxb);
  k_pool_fin<<<cdiv(BB * 64, 256), 256, 0, stream>>>(xsum, gcnt, xmean, xmaxb);
  k_gate_logit<<<cdiv(NN, 256), 256, 0, stream>>>(h, gateW, gateB, batch, glog, gmaxb);
  k_seg_exp<<<cdiv(NN, 256), 256, 0, stream>>>(glog, gmaxb, gden, batch);
  k_seg_scatter64<<<cdiv(NN, 256), 256, 0, stream>>>(h, glog, gden, batch, xatt);

  // Set2Set (3 steps)
  hipMemsetAsync(qstar, 0, BB * 128 * 4, stream);
  hipMemsetAsync(hs, 0, BB * 64 * 4, stream);
  hipMemsetAsync(cs, 0, BB * 64 * 4, stream);
  for (int step = 0; step < 3; ++step) {
    k_lstm_gates<<<cdiv(256 * BB, 256), 256, 0, stream>>>(qstar, hs, Wih, Whh, bih, bhh, gates);
    k_lstm_update<<<cdiv(BB * 64, 256), 256, 0, stream>>>(gates, cs, hs);
    hipMemsetAsync(smaxb, 0, BB * 4, stream);
    hipMemsetAsync(sden, 0, BB * 4, stream);
    hipMemsetAsync(rr, 0, BB * 64 * 4, stream);
    k_s2s_logit<<<cdiv(NN, 256), 256, 0, stream>>>(h, hs, batch, slog, smaxb);
    k_seg_exp<<<cdiv(NN, 256), 256, 0, stream>>>(slog, smaxb, sden, batch);
    k_seg_scatter64<<<cdiv(NN, 256), 256, 0, stream>>>(h, slog, sden, batch, rr);
    k_qstar<<<cdiv(BB * 128, 256), 256, 0, stream>>>(hs, rr, qstar);
  }

  // concat + MLP head
  k_concat<<<cdiv(BB * 384, 256), 256, 0, stream>>>(xsum, (const float*)xmaxb, xmean, xatt, qstar, z);
  k_mlp<<<cdiv(BB * 128, 256), 256, 0, stream>>>(z,  mW[0], mB[0], t1, 384, 128, 1);
  k_mlp<<<cdiv(BB * 64, 256), 256, 0, stream>>>(t1, mW[1], mB[1], t2, 128, 64, 1);
  k_mlp<<<cdiv(BB * 32, 256), 256, 0, stream>>>(t2, mW[2], mB[2], t3, 64, 32, 1);
  k_mlp<<<cdiv(BB, 256), 256, 0, stream>>>(t3, mW[3], mB[3], (float*)d_out, 32, 1, 0);
}
